// SGA_38482906972450
// MI455X (gfx1250) — matrix operand order repaired and checked by reading, not compiled, not run on hardware
//
#include <hip/hip_runtime.h>

// ---------------------------------------------------------------------------
// Types
// ---------------------------------------------------------------------------
typedef __attribute__((ext_vector_type(16))) __bf16 v16bf;
typedef __attribute__((ext_vector_type(8)))  __bf16 v8bf;
typedef __attribute__((ext_vector_type(8)))  float  v8f;
typedef __attribute__((ext_vector_type(4)))  unsigned int u32x4;
typedef __attribute__((ext_vector_type(8)))  int i32x8;
typedef __attribute__((ext_vector_type(4)))  int i32x4;

#if __has_builtin(__builtin_amdgcn_tensor_load_to_lds) && \
    __has_builtin(__builtin_amdgcn_s_wait_tensorcnt)
#define HAVE_TDM 1
#else
#define HAVE_TDM 0
#endif

static __device__ __forceinline__ unsigned short f2bf(float f) {
    union { float f; unsigned u; } x{f};
    unsigned r = x.u + 0x7fffu + ((x.u >> 16) & 1u);   // round-to-nearest-even
    return (unsigned short)(r >> 16);
}

// LDS byte offset of a __shared__ object (addrspace(3) cast)
static __device__ __forceinline__ unsigned ldsoff(const void* p) {
    return (unsigned)(unsigned long long)
        (const __attribute__((address_space(3))) void*)p;
}

// ---- 16-lane butterfly reductions via DPP (no LDS traffic) ----------------
template<int CTRL>
static __device__ __forceinline__ float dppmov(float x) {
    return __builtin_bit_cast(float,
        __builtin_amdgcn_update_dpp(0, __builtin_bit_cast(int, x),
                                    CTRL, 0xF, 0xF, true));
}
static __device__ __forceinline__ float redmax16(float v) {
    v = fmaxf(v, dppmov<0xB1>(v));    // quad_perm xor1
    v = fmaxf(v, dppmov<0x4E>(v));    // quad_perm xor2
    v = fmaxf(v, dppmov<0x141>(v));   // row_half_mirror
    v = fmaxf(v, dppmov<0x140>(v));   // row_mirror
    return v;
}
static __device__ __forceinline__ float redsum16(float v) {
    v += dppmov<0xB1>(v);
    v += dppmov<0x4E>(v);
    v += dppmov<0x141>(v);
    v += dppmov<0x140>(v);
    return v;
}

// ---------------------------------------------------------------------------
// TDM: 2D tile (rows x 32 bf16) global -> LDS, with 16B pad after each 64B row
// (LDS row stride = 80B, matching the padded fragment layout).
// ---------------------------------------------------------------------------
#if HAVE_TDM
static __device__ __forceinline__ void tdm2d(unsigned lds, const void* g,
                                             int rows, int strideElems) {
    unsigned long long ga = (unsigned long long)g;
    u32x4 g0;
    g0[0] = 1u;                                   // count=1, user descriptor
    g0[1] = lds;                                  // lds_addr (bytes)
    g0[2] = (unsigned)ga;                         // global_addr[31:0]
    g0[3] = (unsigned)((ga >> 32) & 0x1FFFFFFu) | (2u << 30);  // addr[56:32], type=2
    i32x8 g1;
    g1[0] = (1 << 16)      // data_size = 2B
          | (1 << 20)      // pad_enable
          | (3 << 22)      // pad_interval: 64B
          | (3 << 25);     // pad_amount: 4 DWORDs = 16B
    g1[1] = (int)(32u << 16);                     // tensor_dim0 = 32
    g1[2] = (int)((unsigned)rows << 16);          // tensor_dim1 = rows
    g1[3] = (int)(32u << 16);                     // tile_dim0 = 32
    g1[4] = rows;                                 // tile_dim1 = rows, tile_dim2 = 0
    g1[5] = strideElems;                          // tensor_dim0_stride lo
    g1[6] = 0;
    g1[7] = 0;
    i32x4 z4 = {0, 0, 0, 0};
    i32x8 z8 = {0, 0, 0, 0, 0, 0, 0, 0};
    __builtin_amdgcn_tensor_load_to_lds(g0, g1, z4, z4, z8, 0);
}
#endif

// ---------------------------------------------------------------------------
// GEMM: Y[M,N] = act( X[M,K] @ W[N,K]^T + bias[N] ),  X/W bf16, Y f32 or bf16
// Requires M%128==0, N%128==0, K%32==0.  256 thr = 8 waves; block tile
// 128x128; wave tile 32x64 (2x4 WMMA frags = 8 WMMAs per wave per K-step).
// Double-buffered LDS; tiles staged by the Tensor Data Mover (TENSORcnt).
// ---------------------------------------------------------------------------
#define BM 128
#define BN 128
#define BK 32
#define LSTR 40   // padded LDS row stride (bf16): 80B

__global__ void __launch_bounds__(256)
gemm_kernel(const unsigned short* __restrict__ X,
            const unsigned short* __restrict__ W,
            const float* __restrict__ bias, void* __restrict__ Yv,
            int M, int N, int K, int act, int out_bf)
{
    __shared__ __align__(16) unsigned short Xs[2][BM][LSTR];
    __shared__ __align__(16) unsigned short Ws[2][BN][LSTR];

    const int tid  = threadIdx.x;
    const int lane = tid & 31;
    const int wave = tid >> 5;
    const int wm   = wave & 3;      // 4 wave-rows  (32 rows each)
    const int wn   = wave >> 2;     // 2 wave-cols  (64 cols each)
    const int m0   = blockIdx.y * BM;
    const int n0   = blockIdx.x * BN;
    const int lm   = lane & 15;
    const int lh   = lane >> 4;
    const int nk   = K >> 5;

    v8f acc[2][4] = {};

#if HAVE_TDM
    if (wave == 0) {
        tdm2d(ldsoff(&Xs[0][0][0]), X + (size_t)m0 * K, BM, K);
        tdm2d(ldsoff(&Ws[0][0][0]), W + (size_t)n0 * K, BN, K);
    }
#endif
    for (int it = 0; it < nk; ++it) {
        const int buf = it & 1;
#if HAVE_TDM
        if (wave == 0) {
            if (it + 1 < nk) {
                const int kk = (it + 1) << 5, nb = (it + 1) & 1;
                tdm2d(ldsoff(&Xs[nb][0][0]), X + (size_t)m0 * K + kk, BM, K);
                tdm2d(ldsoff(&Ws[nb][0][0]), W + (size_t)n0 * K + kk, BN, K);
                __builtin_amdgcn_s_wait_tensorcnt((short)2);  // prev pair done
            } else {
                __builtin_amdgcn_s_wait_tensorcnt((short)0);
            }
        }
#else
        {   // fallback: per-thread staging
            const int k0 = it << 5;
            #pragma unroll
            for (int i = 0; i < 2; ++i) {
                int j = tid + i * 256;              // 512 x 16B chunks
                int r = j >> 2, c8 = (j & 3) * 8;
                *(uint4*)&Xs[buf][r][c8] =
                    *(const uint4*)(X + (size_t)(m0 + r) * K + k0 + c8);
                *(uint4*)&Ws[buf][r][c8] =
                    *(const uint4*)(W + (size_t)(n0 + r) * K + k0 + c8);
            }
        }
#endif
        __syncthreads();

        v16bf a[2], b[4];
        #pragma unroll
        for (int f = 0; f < 2; ++f) {
            const int row = wm * 32 + f * 16 + lm;       // A: lane = M row
            ((v8bf*)&a[f])[0] = *(const v8bf*)&Xs[buf][row][8 * lh];
            ((v8bf*)&a[f])[1] = *(const v8bf*)&Xs[buf][row][16 + 8 * lh];
        }
        #pragma unroll
        for (int f = 0; f < 4; ++f) {
            const int col = wn * 64 + f * 16 + lm;       // B: lane = N col
            ((v8bf*)&b[f])[0] = *(const v8bf*)&Ws[buf][col][8 * lh];
            ((v8bf*)&b[f])[1] = *(const v8bf*)&Ws[buf][col][16 + 8 * lh];
        }
        #pragma unroll
        for (int i = 0; i < 2; ++i)
            #pragma unroll
            for (int j = 0; j < 4; ++j)
                acc[i][j] = __builtin_amdgcn_wmma_f32_16x16x32_bf16(
                    false, a[i], false, b[j], (short)0, acc[i][j], false, false);
        __syncthreads();
    }

    #pragma unroll
    for (int i = 0; i < 2; ++i) {
        #pragma unroll
        for (int j = 0; j < 4; ++j) {
            const int col = n0 + wn * 64 + j * 16 + lm;
            const float bv = bias[col];
            #pragma unroll
            for (int r = 0; r < 8; ++r) {
                const int row = m0 + wm * 32 + i * 16 + r + 8 * lh;
                float v = acc[i][j][r] + bv;
                if (act == 1) v = fmaxf(v, 0.f);
                if (out_bf) ((unsigned short*)Yv)[(size_t)row * N + col] = f2bf(v);
                else        ((float*)Yv)[(size_t)row * N + col] = v;
            }
        }
    }
}

// ---------------------------------------------------------------------------
// Flash attention: one wave per (b, h, 16-row q tile); online softmax; WMMA
// for QK^T and PV; DPP reductions; V pre-transposed for contiguous B-frags.
// ---------------------------------------------------------------------------
__global__ void __launch_bounds__(128)
attn_kernel(const unsigned short* __restrict__ Q,   // [B,512,1024] bf16
            const unsigned short* __restrict__ Kb,  // [B,Skv,1024] bf16
            const unsigned short* __restrict__ Vt,  // [B*16*64, Skv] bf16
            const int* __restrict__ mask,           // [B,Skv] or null
            unsigned short* __restrict__ O,         // [B,512,1024] bf16
            int Skv)
{
    __shared__ __align__(16) unsigned short Pl[4][16][LSTR];

    const int tid  = threadIdx.x;
    const int lane = tid & 31;
    const int wv   = tid >> 5;
    const int gid  = blockIdx.x * 4 + wv;
    const int qt   = gid & 31;
    const int bh   = gid >> 5;
    const int b    = bh >> 4;
    const int h    = bh & 15;
    const int lm   = lane & 15;
    const int lh   = lane >> 4;

    v16bf qa[2];
    {
        const unsigned short* qp =
            Q + ((size_t)(b * 512 + qt * 16 + lm) * 1024) + h * 64;
        #pragma unroll
        for (int c = 0; c < 2; ++c) {
            ((v8bf*)&qa[c])[0] = *(const v8bf*)(qp + 32 * c + 8 * lh);
            ((v8bf*)&qa[c])[1] = *(const v8bf*)(qp + 32 * c + 16 + 8 * lh);
        }
    }

    v8f o[4] = {};
    float mrun[8], lrun[8];
    #pragma unroll
    for (int r = 0; r < 8; ++r) { mrun[r] = -1e30f; lrun[r] = 0.f; }

    for (int k0 = 0; k0 < Skv; k0 += 32) {
        v8f s[2];
        #pragma unroll
        for (int j = 0; j < 2; ++j) {
            const int n = k0 + 16 * j + lm;
            const unsigned short* kp = Kb + ((size_t)(b * Skv + n) * 1024) + h * 64;
            v8f sj = {};
            #pragma unroll
            for (int c = 0; c < 2; ++c) {
                v16bf kb;
                ((v8bf*)&kb)[0] = *(const v8bf*)(kp + 32 * c + 8 * lh);
                ((v8bf*)&kb)[1] = *(const v8bf*)(kp + 32 * c + 16 + 8 * lh);
                sj = __builtin_amdgcn_wmma_f32_16x16x32_bf16(
                    false, qa[c], false, kb, (short)0, sj, false, false);
            }
            bool masked = mask ? (mask[b * Skv + n] == 1) : false;
            #pragma unroll
            for (int r = 0; r < 8; ++r) {
                float v = sj[r] * 0.125f;
                sj[r] = masked ? -1e9f : v;
            }
            s[j] = sj;
        }

        float mnew[8], corr[8];
        #pragma unroll
        for (int r = 0; r < 8; ++r) {
            float v = redmax16(fmaxf(s[0][r], s[1][r]));
            mnew[r] = fmaxf(mrun[r], v);
            corr[r] = __expf(mrun[r] - mnew[r]);
            mrun[r] = mnew[r];
        }
        #pragma unroll
        for (int r = 0; r < 8; ++r) {
            float p0 = __expf(s[0][r] - mnew[r]);
            float p1 = __expf(s[1][r] - mnew[r]);
            s[0][r] = p0; s[1][r] = p1;
            lrun[r] = lrun[r] * corr[r] + redsum16(p0 + p1);
        }
        #pragma unroll
        for (int t = 0; t < 4; ++t)
            #pragma unroll
            for (int r = 0; r < 8; ++r)
                o[t][r] *= corr[r];

        // P: C-layout -> A-layout via per-wave LDS round trip
        #pragma unroll
        for (int r = 0; r < 8; ++r) {
            Pl[wv][r + 8 * lh][lm]      = f2bf(s[0][r]);
            Pl[wv][r + 8 * lh][16 + lm] = f2bf(s[1][r]);
        }
        asm volatile("s_wait_dscnt 0" ::: "memory");
        v16bf pa;
        ((v8bf*)&pa)[0] = *(const v8bf*)&Pl[wv][lm][8 * lh];
        ((v8bf*)&pa)[1] = *(const v8bf*)&Pl[wv][lm][16 + 8 * lh];

        #pragma unroll
        for (int t = 0; t < 4; ++t) {
            const unsigned short* vp =
                Vt + ((size_t)(bh * 64 + 16 * t + lm) * Skv) + k0 + 8 * lh;
            v16bf vb;
            ((v8bf*)&vb)[0] = *(const v8bf*)vp;
            ((v8bf*)&vb)[1] = *(const v8bf*)(vp + 16);
            o[t] = __builtin_amdgcn_wmma_f32_16x16x32_bf16(
                false, pa, false, vb, (short)0, o[t], false, false);
        }
    }

    #pragma unroll
    for (int t = 0; t < 4; ++t) {
        #pragma unroll
        for (int r = 0; r < 8; ++r) {
            const int row = qt * 16 + r + 8 * lh;
            float v = o[t][r] / lrun[r];
            O[((size_t)(b * 512 + row) * 1024) + h * 64 + 16 * t + lm] = f2bf(v);
        }
    }
}

// ---------------------------------------------------------------------------
// f32 -> bf16 conversion (n % 4 == 0)
// ---------------------------------------------------------------------------
__global__ void cvt_kernel(const float* __restrict__ in,
                           unsigned short* __restrict__ out, int n)
{
    int i = (blockIdx.x * 256 + threadIdx.x) * 4;
    if (i >= n) return;
    float4 f = *(const float4*)(in + i);
    unsigned short u[4] = { f2bf(f.x), f2bf(f.y), f2bf(f.z), f2bf(f.w) };
    *(uint2*)(out + i) = *(const uint2*)u;
}

// ---------------------------------------------------------------------------
// V transpose: vb[b,s,h*64+d] (bf16) -> vt[((b*16+h)*64+d), s] (bf16)
// ---------------------------------------------------------------------------
__global__ void vtrans_kernel(const unsigned short* __restrict__ Vb,
                              unsigned short* __restrict__ Vt, int Skv)
{
    int idx = blockIdx.x * 256 + threadIdx.x;
    int total = 16 * 1024 * Skv;
    if (idx >= total) return;
    int s   = idx % Skv;
    int rem = idx / Skv;
    int b   = rem >> 10;
    int hd  = rem & 1023;
    Vt[idx] = Vb[((size_t)(b * Skv + s) << 10) + hd];
}

// ---------------------------------------------------------------------------
// Residual + LayerNorm: out = g*norm(x+y)+b (f32), optional bf16 copy
// ---------------------------------------------------------------------------
__global__ void __launch_bounds__(256)
ln_kernel(const float* __restrict__ X, const float* __restrict__ Y,
          const float* __restrict__ g, const float* __restrict__ bb,
          float* __restrict__ out, unsigned short* __restrict__ outb)
{
    __shared__ float rs[8], rs2[8];
    const int row = blockIdx.x;
    const int tid = threadIdx.x;
    const float* xr = X + (size_t)row * 1024;
    const float* yr = Y + (size_t)row * 1024;

    float v[4], s = 0.f, s2 = 0.f;
    #pragma unroll
    for (int i = 0; i < 4; ++i) {
        float t = xr[tid + 256 * i] + yr[tid + 256 * i];
        v[i] = t; s += t; s2 += t * t;
    }
    s  = redsum16(s);  s  += __shfl_xor(s, 16, 32);
    s2 = redsum16(s2); s2 += __shfl_xor(s2, 16, 32);
    if ((tid & 31) == 0) { rs[tid >> 5] = s; rs2[tid >> 5] = s2; }
    __syncthreads();
    float S = 0.f, S2 = 0.f;
    #pragma unroll
    for (int w = 0; w < 8; ++w) { S += rs[w]; S2 += rs2[w]; }
    const float mean = S * (1.f / 1024.f);
    const float var  = S2 * (1.f / 1024.f) - mean * mean;
    const float rstd = rsqrtf(var + 1e-6f);
    #pragma unroll
    for (int i = 0; i < 4; ++i) {
        int c = tid + 256 * i;
        float o = g[c] * (v[i] - mean) * rstd + bb[c];
        out[(size_t)row * 1024 + c] = o;
        if (outb) outb[(size_t)row * 1024 + c] = f2bf(o);
    }
}

// ---------------------------------------------------------------------------
// Host wrapper
// ---------------------------------------------------------------------------
extern "C" void kernel_launch(void* const* d_in, const int* in_sizes, int n_in,
                              void* d_out, int out_size, void* d_ws, size_t ws_size,
                              hipStream_t stream)
{
    (void)in_sizes; (void)n_in; (void)out_size; (void)ws_size;
    const float* img   = (const float*)d_in[0];
    const float* ques  = (const float*)d_in[1];
    const float* kg    = (const float*)d_in[2];
    const int*   qmask = (const int*)d_in[3];
    const int*   kmask = (const int*)d_in[4];
    const float* aqw = (const float*)d_in[5];  const float* aqb = (const float*)d_in[6];
    const float* akw = (const float*)d_in[7];  const float* akb = (const float*)d_in[8];
    const float* avw = (const float*)d_in[9];  const float* avb = (const float*)d_in[10];
    const float* aow = (const float*)d_in[11]; const float* aob = (const float*)d_in[12];
    const float* w1  = (const float*)d_in[13]; const float* b1  = (const float*)d_in[14];
    const float* w2  = (const float*)d_in[15]; const float* b2  = (const float*)d_in[16];
    const float* lng = (const float*)d_in[17]; const float* lnb = (const float*)d_in[18];

    constexpr size_t MB = 1024ull * 1024ull;
    char* ws = (char*)d_ws;
    float*          x     = (float*)(ws + 0);              // 32 MB f32
    float*          y     = (float*)(ws + 32 * MB);        // 32 MB f32
    unsigned short* xb    = (unsigned short*)(ws + 64 * MB);   // 16 MB bf16
    unsigned short* qbuf  = (unsigned short*)(ws + 80 * MB);   // 16 MB
    unsigned short* kbuf  = (unsigned short*)(ws + 96 * MB);   // 16 MB
    unsigned short* vbuf  = (unsigned short*)(ws + 112 * MB);  // 16 MB
    unsigned short* vt    = (unsigned short*)(ws + 128 * MB);  // 16 MB
    unsigned short* obuf  = (unsigned short*)(ws + 144 * MB);  // 16 MB
    unsigned short* wslot = (unsigned short*)(ws + 160 * MB);  // 8 MB (weights)
    unsigned short* quesb = (unsigned short*)(ws + 168 * MB);  // 2 MB
    unsigned short* kgb   = (unsigned short*)(ws + 170 * MB);  // 4 MB
    unsigned short* hb    = (unsigned short*)(ws + 80 * MB);   // 64 MB (reuse)

    cvt_kernel<<<8192, 256, 0, stream>>>(img, xb, 8388608);
    cvt_kernel<<<1024, 256, 0, stream>>>(ques, quesb, 1048576);
    cvt_kernel<<<2048, 256, 0, stream>>>(kg, kgb, 2097152);

    for (int blk = 0; blk < 3; ++blk) {
        const unsigned short* kvb = (blk == 0) ? xb : (blk == 1 ? quesb : kgb);
        const int Skv = (blk == 0) ? 512 : (blk == 1 ? 64 : 128);
        const int Mkv = 16 * Skv;
        const int* msk = (blk == 0) ? nullptr : (blk == 1 ? qmask : kmask);
        const size_t wo = (size_t)blk * 1024 * 1024;
        const size_t bo = (size_t)blk * 1024;

        cvt_kernel<<<1024, 256, 0, stream>>>(aqw + wo, wslot, 1048576);
        gemm_kernel<<<dim3(8, 64), 256, 0, stream>>>(
            xb, wslot, aqb + bo, qbuf, 8192, 1024, 1024, 0, 1);

        cvt_kernel<<<1024, 256, 0, stream>>>(akw + wo, wslot, 1048576);
        gemm_kernel<<<dim3(8, Mkv / 128), 256, 0, stream>>>(
            kvb, wslot, akb + bo, kbuf, Mkv, 1024, 1024, 0, 1);

        cvt_kernel<<<1024, 256, 0, stream>>>(avw + wo, wslot, 1048576);
        gemm_kernel<<<dim3(8, Mkv / 128), 256, 0, stream>>>(
            kvb, wslot, avb + bo, vbuf, Mkv, 1024, 1024, 0, 1);

        const int tot = 16 * 1024 * Skv;
        vtrans_kernel<<<(tot + 255) / 256, 256, 0, stream>>>(vbuf, vt, Skv);

        attn_kernel<<<2048, 128, 0, stream>>>(qbuf, kbuf, vt, msk, obuf, Skv);

        cvt_kernel<<<1024, 256, 0, stream>>>(aow + wo, wslot, 1048576);
        gemm_kernel<<<dim3(8, 64), 256, 0, stream>>>(
            obuf, wslot, aob + bo, y, 8192, 1024, 1024, 0, 0);

        ln_kernel<<<8192, 256, 0, stream>>>(
            (blk == 0) ? img : x, y, lng + bo, lnb + bo, x, xb);
    }

    // FFN
    cvt_kernel<<<4096, 256, 0, stream>>>(w1, wslot, 4194304);
    gemm_kernel<<<dim3(32, 64), 256, 0, stream>>>(
        xb, wslot, b1, hb, 8192, 4096, 1024, 1, 1);
    cvt_kernel<<<4096, 256, 0, stream>>>(w2, wslot, 4194304);
    gemm_kernel<<<dim3(8, 64), 256, 0, stream>>>(
        hb, wslot, b2, y, 8192, 1024, 4096, 0, 0);
    ln_kernel<<<8192, 256, 0, stream>>>(x, y, lng + 3 * 1024, lnb + 3 * 1024,
                                        (float*)d_out, nullptr);
}
